// PlugAndPlayDCN_6201932775437
// MI455X (gfx1250) — matrix operand order repaired and checked by reading, not compiled, not run on hardware
//
#include <hip/hip_runtime.h>
#include <hip/hip_bf16.h>

// Problem constants (from reference setup_inputs)
#define BATCH 4
#define CIN   256
#define COUT  256
#define HDIM  64
#define WDIM  64
#define K2    9
#define KDIM  2304      // CIN * K2 (GEMM reduction dim)
#define NPIX  16384     // BATCH * HDIM * WDIM (GEMM N dim)

typedef __bf16        v16bf __attribute__((ext_vector_type(16)));
typedef float         v8f   __attribute__((ext_vector_type(8)));
typedef unsigned int  v4u   __attribute__((ext_vector_type(4)));

typedef union { v4u u[2]; v16bf v; } Frag;

__device__ __forceinline__ unsigned short f2bf(float f) {
    unsigned int u = __float_as_uint(f);
    u += 0x7FFFu + ((u >> 16) & 1u);      // round-to-nearest-even
    return (unsigned short)(u >> 16);
}

// A fragment: per-lane K runs {0-7} and {16-23} (+8 for upper half-wave)
__device__ __forceinline__ void loadA(Frag& f, const unsigned short* p) {
    f.u[0] = *(const v4u*)(p);
    f.u[1] = *(const v4u*)(p + 16);
}
// B fragment: per-lane K runs {0-7} and {16-23} (+8 for upper half-wave)
__device__ __forceinline__ void loadB(Frag& f, const unsigned short* p) {
    f.u[0] = *(const v4u*)(p);
    f.u[1] = *(const v4u*)(p + 16);
}

// ---------------------------------------------------------------------------
// Kernel 1: convert conv weight (Cout,Cin,3,3) f32 -> bf16 (K-contiguous)
// ---------------------------------------------------------------------------
__global__ __launch_bounds__(256) void pack_weight_bf16(const float* __restrict__ w,
                                                        unsigned short* __restrict__ a) {
    int i = blockIdx.x * 256 + threadIdx.x;   // 0 .. COUT*KDIM-1
    a[i] = f2bf(w[i]);
}

// ---------------------------------------------------------------------------
// Kernel 2: offset/mask conv: pout[p*27+ch] = conv3x3(x, p_weight)[ch] + p_bias[ch]
// ---------------------------------------------------------------------------
__global__ __launch_bounds__(256) void offset_conv(const float* __restrict__ x,
                                                   const float* __restrict__ pw,
                                                   const float* __restrict__ pb,
                                                   float* __restrict__ pout) {
    int gid = blockIdx.x * 256 + threadIdx.x;   // NPIX*27 threads
    int p  = gid / 27;
    int ch = gid - p * 27;
    int b  = p >> 12;
    int hw = p & 4095;
    int y  = hw >> 6;
    int xc = hw & 63;
    float acc = pb[ch];
    const float* xb  = x  + (size_t)b * CIN * 4096;
    const float* wch = pw + (size_t)ch * KDIM;
    for (int c = 0; c < CIN; ++c) {
        const float* xcp = xb + c * 4096;
        const float* wc  = wch + c * 9;
        #pragma unroll
        for (int ky = 0; ky < 3; ++ky) {
            int yy = y + ky - 1;
            if (yy < 0 || yy > 63) continue;
            const float* row = xcp + yy * 64;
            #pragma unroll
            for (int kx = 0; kx < 3; ++kx) {
                int xx = xc + kx - 1;
                if (xx < 0 || xx > 63) continue;
                acc += row[xx] * wc[ky * 3 + kx];
            }
        }
    }
    pout[gid] = acc;
}

// ---------------------------------------------------------------------------
// Kernel 3: bilinear sample + mask -> S[p][c*9+kk] in bf16 (pixel-major, K contig)
// ---------------------------------------------------------------------------
__global__ __launch_bounds__(256) void sample_kernel(const float* __restrict__ x,
                                                     const float* __restrict__ pout,
                                                     unsigned short* __restrict__ S) {
    int p  = blockIdx.x;
    int b  = p >> 12;
    int hw = p & 4095;
    int y  = hw >> 6;
    int xc = hw & 63;
    __shared__ int   s_r0[K2], s_r1[K2], s_c0[K2], s_c1[K2];
    __shared__ float s_w00[K2], s_w01[K2], s_w10[K2], s_w11[K2];
    int tid = threadIdx.x;
    if (tid < K2) {
        const float* pp = pout + (size_t)p * 27;
        float dy = pp[2 * tid];
        float dx = pp[2 * tid + 1];
        float m  = 1.f / (1.f + __expf(-pp[18 + tid]));
        float py = (float)(y  - 1 + tid / 3) + dy;
        float px = (float)(xc - 1 + tid % 3) + dx;
        float y0f = floorf(py), x0f = floorf(px);
        float ay = py - y0f,  ax = px - x0f;
        int y0 = (int)y0f, x0 = (int)x0f;
        int y1 = y0 + 1,   x1 = x0 + 1;
        float vy0 = (y0 >= 0 && y0 < HDIM) ? 1.f : 0.f;
        float vy1 = (y1 >= 0 && y1 < HDIM) ? 1.f : 0.f;
        float vx0 = (x0 >= 0 && x0 < WDIM) ? 1.f : 0.f;
        float vx1 = (x1 >= 0 && x1 < WDIM) ? 1.f : 0.f;
        s_w00[tid] = (1.f - ay) * (1.f - ax) * vy0 * vx0 * m;
        s_w01[tid] = (1.f - ay) * ax         * vy0 * vx1 * m;
        s_w10[tid] = ay         * (1.f - ax) * vy1 * vx0 * m;
        s_w11[tid] = ay         * ax         * vy1 * vx1 * m;
        s_r0[tid] = min(max(y0, 0), HDIM - 1) * WDIM;
        s_r1[tid] = min(max(y1, 0), HDIM - 1) * WDIM;
        s_c0[tid] = min(max(x0, 0), WDIM - 1);
        s_c1[tid] = min(max(x1, 0), WDIM - 1);
    }
    __syncthreads();
    const float* xb = x + (size_t)b * CIN * 4096;
    unsigned short* Sp = S + (size_t)p * KDIM;
    #pragma unroll
    for (int it = 0; it < 9; ++it) {
        int idx = it * 256 + tid;         // 0..2303, coalesced store
        int c   = idx / 9;
        int kk  = idx - c * 9;
        const float* xcp = xb + c * 4096;
        int r0 = s_r0[kk], r1 = s_r1[kk];
        float v = s_w00[kk] * xcp[r0 + s_c0[kk]]
                + s_w01[kk] * xcp[r0 + s_c1[kk]]
                + s_w10[kk] * xcp[r1 + s_c0[kk]]
                + s_w11[kk] * xcp[r1 + s_c1[kk]];
        Sp[idx] = f2bf(v);
    }
}

// ---------------------------------------------------------------------------
// Kernel 4: WMMA GEMM  out(M=256 x N=16384) = A(256 x 2304) * S^T + bias
// One wave -> 32(M) x 64(N). Explicit ping/pong double buffering of all
// fragments: next K-step's 12 b128 loads are issued before the current
// 8 wmma consume their set, so waits overlap load latency with WMMA issue.
// ---------------------------------------------------------------------------
__global__ __launch_bounds__(256) void dcn_wmma_gemm(const unsigned short* __restrict__ A,
                                                     const unsigned short* __restrict__ S,
                                                     const float* __restrict__ bias,
                                                     float* __restrict__ out) {
    int lane = threadIdx.x & 31;
    int wave = threadIdx.x >> 5;
    int lt = blockIdx.x * 8 + wave;       // 0..2047 wave-tiles
    int tile_m = lt & 7;                  // 8 tiles of 32 M rows
    int tile_n = lt >> 3;                 // 256 super-tiles of 64 pixels
    int lm = lane & 15;
    int hi = lane >> 4;                   // half-wave selector

    const unsigned short* Arow0 = A + (size_t)((tile_m << 5) + lm) * KDIM + hi * 8;
    const unsigned short* Arow1 = Arow0 + (size_t)16 * KDIM;
    int n0 = (tile_n << 6) + lm;
    const unsigned short* Bp[4];
    #pragma unroll
    for (int f = 0; f < 4; ++f)
        Bp[f] = S + (size_t)(n0 + f * 16) * KDIM + hi * 8;

    v8f acc[8];                           // acc[mh*4 + f]
    #pragma unroll
    for (int mh = 0; mh < 2; ++mh) {
        #pragma unroll
        for (int r = 0; r < 8; ++r) {
            float bv = bias[(tile_m << 5) + (mh << 4) + (hi << 3) + r];
            acc[mh * 4 + 0][r] = bv;
            acc[mh * 4 + 1][r] = bv;
            acc[mh * 4 + 2][r] = bv;
            acc[mh * 4 + 3][r] = bv;
        }
    }

    Frag a0[2], a1[2], bf[2][4];
    // Prologue: fill set 0 at kb=0
    loadA(a0[0], Arow0);
    loadA(a1[0], Arow1);
    #pragma unroll
    for (int f = 0; f < 4; ++f) loadB(bf[0][f], Bp[f]);

    // KDIM = 64 * 36 -> exact unroll-by-2 ping/pong
    for (int kb = 0; kb < KDIM; kb += 64) {
        // Prefetch set 1 @ kb+32
        loadA(a0[1], Arow0 + kb + 32);
        loadA(a1[1], Arow1 + kb + 32);
        #pragma unroll
        for (int f = 0; f < 4; ++f) loadB(bf[1][f], Bp[f] + kb + 32);
        // Compute set 0
        #pragma unroll
        for (int f = 0; f < 4; ++f) {
            acc[f] = __builtin_amdgcn_wmma_f32_16x16x32_bf16(
                false, a0[0].v, false, bf[0][f].v, (short)0, acc[f], false, false);
            acc[4 + f] = __builtin_amdgcn_wmma_f32_16x16x32_bf16(
                false, a1[0].v, false, bf[0][f].v, (short)0, acc[4 + f], false, false);
        }
        // Prefetch set 0 @ kb+64 (guarded: stay inside A / S allocations)
        if (kb + 64 < KDIM) {
            loadA(a0[0], Arow0 + kb + 64);
            loadA(a1[0], Arow1 + kb + 64);
            #pragma unroll
            for (int f = 0; f < 4; ++f) loadB(bf[0][f], Bp[f] + kb + 64);
        }
        // Compute set 1
        #pragma unroll
        for (int f = 0; f < 4; ++f) {
            acc[f] = __builtin_amdgcn_wmma_f32_16x16x32_bf16(
                false, a0[1].v, false, bf[1][f].v, (short)0, acc[f], false, false);
            acc[4 + f] = __builtin_amdgcn_wmma_f32_16x16x32_bf16(
                false, a1[1].v, false, bf[1][f].v, (short)0, acc[4 + f], false, false);
        }
    }

    // Store: D layout -> lane N = lm, VGPR r -> M = r + 8*hi. Output is NCHW.
    #pragma unroll
    for (int f = 0; f < 4; ++f) {
        int p  = n0 + f * 16;
        int bb = p >> 12;
        int hw = p & 4095;
        float* obase = out + (((size_t)bb * COUT) << 12) + hw;
        #pragma unroll
        for (int mh = 0; mh < 2; ++mh) {
            #pragma unroll
            for (int r = 0; r < 8; ++r) {
                int o = (tile_m << 5) + (mh << 4) + (hi << 3) + r;
                obase[(size_t)o << 12] = acc[mh * 4 + f][r];
            }
        }
    }
}

// ---------------------------------------------------------------------------
extern "C" void kernel_launch(void* const* d_in, const int* in_sizes, int n_in,
                              void* d_out, int out_size, void* d_ws, size_t ws_size,
                              hipStream_t stream) {
    const float* x    = (const float*)d_in[0];
    const float* w    = (const float*)d_in[1];
    const float* bias = (const float*)d_in[2];
    const float* pw   = (const float*)d_in[3];
    const float* pb   = (const float*)d_in[4];
    float* out = (float*)d_out;

    char* ws = (char*)d_ws;
    unsigned short* Abf = (unsigned short*)ws;                      //  1,179,648 B
    float* pout = (float*)(ws + 1179648);                           //  1,769,472 B
    unsigned short* S = (unsigned short*)(ws + 1179648 + 1769472);  // 75,497,472 B

    pack_weight_bf16<<<(COUT * KDIM) / 256, 256, 0, stream>>>(w, Abf);
    offset_conv<<<(NPIX * 27) / 256, 256, 0, stream>>>(x, pw, pb, pout);
    sample_kernel<<<NPIX, 256, 0, stream>>>(x, pout, S);
    dcn_wmma_gemm<<<256, 256, 0, stream>>>(Abf, S, bias, out);
}
